// TiledFlashLinearAttention_76802605187528
// MI455X (gfx1250) — matrix operand order repaired and checked by reading, not compiled, not run on hardware
//
#include <hip/hip_runtime.h>

// ---------------------------------------------------------------------------
// TFLA forward for MI455X (gfx1250, wave32, WMMA bf16 -> f32 accumulate)
//   B=4, L=4096, DIM=1024, H=16, DH=64, CHUNK=256, Nc=16
// Pipeline:
//   1) proj_gemm  x3 : q/k/v = act(X @ W)      -> bf16, layout [B*H, L, DH]
//   2) chunk_kv       : KVc = kc^T @ vc, zc=colsum(kc)  per (b,h,chunk)
//   3) prefix_scan    : exclusive prefix over chunks (serial dep of the scan)
//   4) attn_chunk     : masked scores, num (intra+inter), den, out  (all WMMA)
//   5) out_gemm       : d_out = att @ Wo  (f32 out)
// Workspace: 4 bf16 tensors (33.5MB each) + KV prefix f32 (16.7MB) + z (256KB)
// ---------------------------------------------------------------------------

typedef __attribute__((ext_vector_type(16))) __bf16 v16bf_t;
typedef __attribute__((ext_vector_type(8)))  float  v8f_t;

#define B_     4
#define L_     4096
#define DIM_   1024
#define H_     16
#define DH_    64
#define CHUNK_ 256
#define NC_    16
#define EPS_   1e-12f

union FragB { v16bf_t v; unsigned int u[8]; unsigned short s[16]; };

__device__ __forceinline__ unsigned short f2bf(float f) {
  union { float f; unsigned int u; } x; x.f = f;
  unsigned int u = x.u;
  u += 0x7fffu + ((u >> 16) & 1u);   // round-to-nearest-even
  return (unsigned short)(u >> 16);
}
__device__ __forceinline__ float bf2f(unsigned short s) {
  union { unsigned int u; float f; } x; x.u = ((unsigned int)s) << 16; return x.f;
}
__device__ __forceinline__ v8f_t zero8() {
  v8f_t z; for (int i = 0; i < 8; ++i) z[i] = 0.f; return z;
}
__device__ __forceinline__ v8f_t wmma_bf16(v16bf_t a, v16bf_t b, v8f_t c) {
  return __builtin_amdgcn_wmma_f32_16x16x32_bf16(false, a, false, b, (short)0, c,
                                                 false, false);
}
// CDNA5 16-bit fragment K-pair bases (CDNA5 ISA 7.12.2)
__device__ __forceinline__ int akb(int v, int hi) { return 2*v + (v >= 4 ? 8 : 0) + 8*hi; }
__device__ __forceinline__ int bkb(int v, int hi) { return 2*v + (v >= 4 ? 8 : 0) + 8*hi; }

// ---------------------------------------------------------------------------
// Kernel 1: projection GEMM.  X[f32, 16384x1024] @ W[f32, 1024x1024]
//   act=1 -> elu(x)+1 ( = x+1 if x>0 else exp(x) ),  act=0 -> identity
//   output bf16 scattered to [B, H, L, DH]
// Tile 128x128x32, 8 waves (2 M x 4 N), wave tile 64x32 = 4x2 WMMA tiles.
// ---------------------------------------------------------------------------
__global__ __launch_bounds__(256) void proj_gemm(
    const float* __restrict__ X, const float* __restrict__ W,
    unsigned short* __restrict__ Y, int act)
{
  __shared__ unsigned short As[128 * 32];  // [m][k] row-major bf16
  __shared__ unsigned short Bs[128 * 32];  // transposed: [n][k] bf16
  const int tid = threadIdx.x, lane = tid & 31, wave = tid >> 5;
  const int hi = lane >> 4, ln = lane & 15;
  const int bm = blockIdx.y * 128, bn = blockIdx.x * 128;
  const int waveM = wave & 1, waveN = wave >> 1;

  v8f_t acc[4][2];
  for (int i = 0; i < 4; ++i) for (int j = 0; j < 2; ++j) acc[i][j] = zero8();

  const int ar = tid >> 1,  ac0 = (tid & 1) * 16;   // A: 2 thr/row, 16 f32 each
  const int bkr = tid >> 3, bc0 = (tid & 7) * 16;   // B: 8 thr/row, 16 f32 each

  for (int kk = 0; kk < DIM_; kk += 32) {
    const float4* ap = (const float4*)(X + (size_t)(bm + ar) * DIM_ + kk + ac0);
#pragma unroll
    for (int q4 = 0; q4 < 4; ++q4) {
      float4 f = ap[q4]; int c = ac0 + q4 * 4;
      As[ar*32 + c + 0] = f2bf(f.x); As[ar*32 + c + 1] = f2bf(f.y);
      As[ar*32 + c + 2] = f2bf(f.z); As[ar*32 + c + 3] = f2bf(f.w);
    }
    const float4* bp = (const float4*)(W + (size_t)(kk + bkr) * DIM_ + bn + bc0);
#pragma unroll
    for (int q4 = 0; q4 < 4; ++q4) {
      float4 f = bp[q4]; int c = bc0 + q4 * 4;
      Bs[(c+0)*32 + bkr] = f2bf(f.x); Bs[(c+1)*32 + bkr] = f2bf(f.y);
      Bs[(c+2)*32 + bkr] = f2bf(f.z); Bs[(c+3)*32 + bkr] = f2bf(f.w);
    }
    __syncthreads();

    FragB af[4], bf[2];
#pragma unroll
    for (int mt = 0; mt < 4; ++mt) {
      int row = waveM * 64 + mt * 16 + ln;
#pragma unroll
      for (int v = 0; v < 8; ++v)
        af[mt].u[v] = *(const unsigned int*)&As[row * 32 + akb(v, hi)];
    }
#pragma unroll
    for (int nt = 0; nt < 2; ++nt) {
      int col = waveN * 32 + nt * 16 + ln;
#pragma unroll
      for (int v = 0; v < 8; ++v)
        bf[nt].u[v] = *(const unsigned int*)&Bs[col * 32 + bkb(v, hi)];
    }
#pragma unroll
    for (int mt = 0; mt < 4; ++mt)
#pragma unroll
      for (int nt = 0; nt < 2; ++nt)
        acc[mt][nt] = wmma_bf16(af[mt].v, bf[nt].v, acc[mt][nt]);
    __syncthreads();
  }

#pragma unroll
  for (int mt = 0; mt < 4; ++mt)
#pragma unroll
    for (int nt = 0; nt < 2; ++nt)
#pragma unroll
      for (int r = 0; r < 8; ++r) {
        float val = acc[mt][nt][r];
        if (act) val = (val > 0.f) ? (val + 1.f) : __expf(val);
        int gr = bm + waveM * 64 + mt * 16 + r + 8 * hi;   // token row
        int gc = bn + waveN * 32 + nt * 16 + ln;           // feature col
        int b = gr >> 12, l = gr & (L_ - 1);
        int h = gc >> 6,  d = gc & (DH_ - 1);
        Y[(((size_t)(b * H_ + h)) * L_ + l) * DH_ + d] = f2bf(val);
      }
}

// ---------------------------------------------------------------------------
// Kernel 2: per-chunk KVc = kc^T @ vc  (64x64, K=256) and zc = colsum(kc)
// One block per (b,h,chunk). 8 waves, 2 tiles each (mt, mt+2) x nt.
// ---------------------------------------------------------------------------
__global__ __launch_bounds__(256) void chunk_kv(
    const unsigned short* __restrict__ kf, const unsigned short* __restrict__ vv,
    float* __restrict__ kvc, float* __restrict__ zc)
{
  extern __shared__ unsigned char smem[];
  unsigned short* kc = (unsigned short*)smem;      // [256][64] row-major
  unsigned short* vT = kc + CHUNK_ * DH_;          // [64][256]  (vT[e][l])
  const int tid = threadIdx.x, lane = tid & 31, wave = tid >> 5;
  const int hi = lane >> 4, ln = lane & 15;
  const int idx = blockIdx.x;                      // bh*NC_ + chunk
  const size_t base = (size_t)idx * CHUNK_ * DH_;  // == (bh*L + c*CHUNK)*DH

  { const unsigned int* src = (const unsigned int*)(kf + base);
    unsigned int* dst = (unsigned int*)kc;
    for (int i = 0; i < 32; ++i) dst[tid + i * 256] = src[tid + i * 256]; }
  for (int i = 0; i < 64; ++i) {
    int e = tid + i * 256; int l = e >> 6, d = e & 63;
    vT[d * CHUNK_ + l] = vv[base + e];
  }
  __syncthreads();

  const int mt0 = wave >> 2, nt = wave & 3;
  v8f_t acc0 = zero8(), acc1 = zero8();
  for (int k0 = 0; k0 < CHUNK_; k0 += 32) {
    FragB b0;
#pragma unroll
    for (int v = 0; v < 8; ++v)
      b0.u[v] = *(const unsigned int*)&vT[(nt * 16 + ln) * CHUNK_ + k0 + bkb(v, hi)];
    FragB a0, a1;
#pragma unroll
    for (int v = 0; v < 8; ++v) {               // A = kc^T : gather stride DH
      int kb = k0 + akb(v, hi);
      unsigned int lo  = kc[(kb    ) * DH_ + mt0 * 16 + ln];
      unsigned int h16 = kc[(kb + 1) * DH_ + mt0 * 16 + ln];
      a0.u[v] = lo | (h16 << 16);
      lo  = kc[(kb    ) * DH_ + (mt0 + 2) * 16 + ln];
      h16 = kc[(kb + 1) * DH_ + (mt0 + 2) * 16 + ln];
      a1.u[v] = lo | (h16 << 16);
    }
    acc0 = wmma_bf16(a0.v, b0.v, acc0);
    acc1 = wmma_bf16(a1.v, b0.v, acc1);
  }
  float* dst = kvc + (size_t)idx * DH_ * DH_;
#pragma unroll
  for (int r = 0; r < 8; ++r) {
    int m = r + 8 * hi, n = nt * 16 + ln;
    dst[(mt0 * 16 + m) * DH_ + n]       = acc0[r];
    dst[((mt0 + 2) * 16 + m) * DH_ + n] = acc1[r];
  }
  if (tid < DH_) {
    float s = 0.f;
    for (int l = 0; l < CHUNK_; ++l) s += bf2f(kc[l * DH_ + tid]);
    zc[(size_t)idx * DH_ + tid] = s;
  }
}

// ---------------------------------------------------------------------------
// Kernel 3: in-place exclusive prefix over the 16 chunks per (b,h)
// ---------------------------------------------------------------------------
__global__ __launch_bounds__(256) void prefix_scan(float* kvc, float* zc)
{
  const int bh = blockIdx.x, tid = threadIdx.x;
  for (int i = 0; i < 16; ++i) {
    int e = tid + i * 256;
    float run = 0.f;
    for (int c = 0; c < NC_; ++c) {
      size_t a = ((size_t)(bh * NC_ + c)) * DH_ * DH_ + e;
      float t = kvc[a]; kvc[a] = run; run += t;
    }
  }
  if (tid < DH_) {
    float run = 0.f;
    for (int c = 0; c < NC_; ++c) {
      size_t a = ((size_t)(bh * NC_ + c)) * DH_ + tid;
      float t = zc[a]; zc[a] = run; run += t;
    }
  }
}

// ---------------------------------------------------------------------------
// Kernel 4: per-chunk attention output.
//   scores = (qc @ kc^T) * tril ; num = scores@vc + qc@KVpre ;
//   den = rowsum(scores) + qc@zpre ; out = num / max(den, EPS)
// One block per (b,h,chunk). Each wave owns 16-row strips (2 iterations).
// den computed with WMMA via a ones-in-column-0 B fragment.
// ---------------------------------------------------------------------------
__global__ __launch_bounds__(256) void attn_chunk(
    const unsigned short* __restrict__ qf, const unsigned short* __restrict__ kf,
    const unsigned short* __restrict__ vv, const float* __restrict__ kvc,
    const float* __restrict__ zc, unsigned short* __restrict__ outY)
{
  extern __shared__ unsigned char smem[];
  unsigned short* kc  = (unsigned short*)smem;        // [256][64]
  unsigned short* vT  = kc + CHUNK_ * DH_;            // [64][256]
  unsigned short* kvT = vT + DH_ * CHUNK_;            // [64][64]   kvT[e][d]
  float*          zsh = (float*)(kvT + DH_ * DH_);    // [64]
  unsigned short* sst = (unsigned short*)(zsh + DH_); // [8][16*32] per-wave stage
  const int tid = threadIdx.x, lane = tid & 31, wave = tid >> 5;
  const int hi = lane >> 4, ln = lane & 15;
  const int idx = blockIdx.x, bh = idx >> 4, c = idx & 15;
  const size_t base = (size_t)idx * CHUNK_ * DH_;

  { const unsigned int* src = (const unsigned int*)(kf + base);
    unsigned int* dst = (unsigned int*)kc;
    for (int i = 0; i < 32; ++i) dst[tid + i * 256] = src[tid + i * 256]; }
  for (int i = 0; i < 64; ++i) {
    int e = tid + i * 256; int l = e >> 6, d = e & 63;
    vT[d * CHUNK_ + l] = vv[base + e];
  }
  { const float* src = kvc + (size_t)idx * DH_ * DH_;
    for (int i = 0; i < 16; ++i) {
      int e = tid + i * 256; int d = e >> 6, ee = e & 63;
      kvT[ee * DH_ + d] = f2bf(src[e]);
    } }
  if (tid < DH_) zsh[tid] = zc[(size_t)idx * DH_ + tid];
  __syncthreads();

  unsigned short* sw = sst + wave * 512;
  for (int it = 0; it < 2; ++it) {
    const int strip = it * 8 + wave;
    const int m0 = strip * 16;
    FragB aq[2];
#pragma unroll
    for (int ks = 0; ks < 2; ++ks)
#pragma unroll
      for (int v = 0; v < 8; ++v)
        aq[ks].u[v] = *(const unsigned int*)(qf + base +
                        (size_t)(m0 + ln) * DH_ + ks * 32 + akb(v, hi));

    v8f_t accn[4]; for (int j = 0; j < 4; ++j) accn[j] = zero8();
    v8f_t accd = zero8();

    // ---- inter-chunk: qc @ KVprefix and qc @ z (den column) ----
#pragma unroll
    for (int ks = 0; ks < 2; ++ks) {
      FragB bz;
#pragma unroll
      for (int v = 0; v < 8; ++v) {
        int kb = ks * 32 + bkb(v, hi);
        unsigned int p = ((unsigned int)f2bf(zsh[kb])) |
                         (((unsigned int)f2bf(zsh[kb + 1])) << 16);
        bz.u[v] = (ln == 0) ? p : 0u;
      }
      accd = wmma_bf16(aq[ks].v, bz.v, accd);
#pragma unroll
      for (int j = 0; j < 4; ++j) {
        FragB bk;
#pragma unroll
        for (int v = 0; v < 8; ++v)
          bk.u[v] = *(const unsigned int*)&kvT[(j * 16 + ln) * DH_ +
                                               ks * 32 + bkb(v, hi)];
        accn[j] = wmma_bf16(aq[ks].v, bk.v, accn[j]);
      }
    }

    // ---- intra-chunk causal: process score tiles in K-pairs of 32 ----
    const int ntmax = strip;   // last (diagonal) 16-col tile index
    for (int nt = 0; nt <= ntmax; nt += 2) {
#pragma unroll
      for (int p = 0; p < 2; ++p) {
        const int tcol = nt + p;
        v8f_t s = zero8();
        if (tcol <= ntmax) {                 // uniform per wave -> EXEC stays full
          const int n0 = tcol * 16;
#pragma unroll
          for (int ks = 0; ks < 2; ++ks) {
            FragB bk;                        // B = kc^T: (k,n) -> kc[n][k]
#pragma unroll
            for (int v = 0; v < 8; ++v)
              bk.u[v] = *(const unsigned int*)&kc[(n0 + ln) * DH_ +
                                                  ks * 32 + bkb(v, hi)];
            s = wmma_bf16(aq[ks].v, bk.v, s);
          }
        }
#pragma unroll
        for (int r = 0; r < 8; ++r) {        // mask + restage as bf16
          int m = r + 8 * hi;
          float val = s[r];
          if (tcol > ntmax) val = 0.f;
          else if (tcol == strip && ln > m) val = 0.f;   // diagonal tril
          sw[m * 32 + p * 16 + ln] = f2bf(val);
        }
      }
      asm volatile("s_wait_dscnt 0" ::: "memory");  // LDS RAW before frag reload
      FragB as;
#pragma unroll
      for (int v = 0; v < 8; ++v)
        as.u[v] = *(const unsigned int*)&sw[ln * 32 + akb(v, hi)];
      const int k0v = nt * 16;
#pragma unroll
      for (int j = 0; j < 4; ++j) {
        FragB bv;
#pragma unroll
        for (int v = 0; v < 8; ++v)
          bv.u[v] = *(const unsigned int*)&vT[(j * 16 + ln) * CHUNK_ +
                                              k0v + bkb(v, hi)];
        accn[j] = wmma_bf16(as.v, bv.v, accn[j]);
      }
      FragB bo;                              // ones in column n==0 -> row sums
#pragma unroll
      for (int v = 0; v < 8; ++v) bo.u[v] = (ln == 0) ? 0x3F803F80u : 0u;
      accd = wmma_bf16(as.v, bo.v, accd);
      asm volatile("s_wait_dscnt 0" ::: "memory");  // WAR before next restage
    }

    // ---- normalize and store out[B, L, H*DH] as bf16 ----
    const int b = bh >> 4, h = bh & 15;
#pragma unroll
    for (int r = 0; r < 8; ++r) {
      float den = __shfl(accd[r], hi * 16, 32);   // den lives at n==0 lane
      den = fmaxf(den, EPS_);
      int m = r + 8 * hi;
      size_t gl = (size_t)b * L_ + c * CHUNK_ + m0 + m;
#pragma unroll
      for (int j = 0; j < 4; ++j) {
        float val = accn[j][r] / den;
        outY[gl * DIM_ + h * DH_ + j * 16 + ln] = f2bf(val);
      }
    }
  }
}

// ---------------------------------------------------------------------------
// Kernel 5: output GEMM.  att[bf16, 16384x1024] @ Wo[f32] -> d_out f32
// ---------------------------------------------------------------------------
__global__ __launch_bounds__(256) void out_gemm(
    const unsigned short* __restrict__ Abf, const float* __restrict__ W,
    float* __restrict__ Y)
{
  __shared__ unsigned short As[128 * 32];
  __shared__ unsigned short Bs[128 * 32];
  const int tid = threadIdx.x, lane = tid & 31, wave = tid >> 5;
  const int hi = lane >> 4, ln = lane & 15;
  const int bm = blockIdx.y * 128, bn = blockIdx.x * 128;
  const int waveM = wave & 1, waveN = wave >> 1;

  v8f_t acc[4][2];
  for (int i = 0; i < 4; ++i) for (int j = 0; j < 2; ++j) acc[i][j] = zero8();

  const int ar = tid >> 1,  ac0 = (tid & 1) * 16;
  const int bkr = tid >> 3, bc0 = (tid & 7) * 16;

  for (int kk = 0; kk < DIM_; kk += 32) {
    { const unsigned int* ap =
        (const unsigned int*)(Abf + (size_t)(bm + ar) * DIM_ + kk + ac0);
      unsigned int* ad = (unsigned int*)&As[ar * 32 + ac0];
#pragma unroll
      for (int i = 0; i < 8; ++i) ad[i] = ap[i]; }
    const float4* bp = (const float4*)(W + (size_t)(kk + bkr) * DIM_ + bn + bc0);
#pragma unroll
    for (int q4 = 0; q4 < 4; ++q4) {
      float4 f = bp[q4]; int c = bc0 + q4 * 4;
      Bs[(c+0)*32 + bkr] = f2bf(f.x); Bs[(c+1)*32 + bkr] = f2bf(f.y);
      Bs[(c+2)*32 + bkr] = f2bf(f.z); Bs[(c+3)*32 + bkr] = f2bf(f.w);
    }
    __syncthreads();

    FragB af[4], bf[2];
#pragma unroll
    for (int mt = 0; mt < 4; ++mt) {
      int row = waveM * 64 + mt * 16 + ln;
#pragma unroll
      for (int v = 0; v < 8; ++v)
        af[mt].u[v] = *(const unsigned int*)&As[row * 32 + akb(v, hi)];
    }
#pragma unroll
    for (int nt = 0; nt < 2; ++nt) {
      int col = waveN * 32 + nt * 16 + ln;
#pragma unroll
      for (int v = 0; v < 8; ++v)
        bf[nt].u[v] = *(const unsigned int*)&Bs[col * 32 + bkb(v, hi)];
    }
#pragma unroll
    for (int mt = 0; mt < 4; ++mt)
#pragma unroll
      for (int nt = 0; nt < 2; ++nt)
        acc[mt][nt] = wmma_bf16(af[mt].v, bf[nt].v, acc[mt][nt]);
    __syncthreads();
  }

#pragma unroll
  for (int mt = 0; mt < 4; ++mt)
#pragma unroll
    for (int nt = 0; nt < 2; ++nt)
#pragma unroll
      for (int r = 0; r < 8; ++r) {
        int gr = bm + waveM * 64 + mt * 16 + r + 8 * hi;
        int gc = bn + waveN * 32 + nt * 16 + ln;
        Y[(size_t)gr * DIM_ + gc] = acc[mt][nt][r];
      }
}

// ---------------------------------------------------------------------------
// Host launcher (graph-capture safe: only kernel launches on `stream`)
// ---------------------------------------------------------------------------
extern "C" void kernel_launch(void* const* d_in, const int* in_sizes, int n_in,
                              void* d_out, int out_size, void* d_ws, size_t ws_size,
                              hipStream_t stream) {
  (void)in_sizes; (void)n_in; (void)out_size; (void)ws_size;
  const float* query = (const float*)d_in[0];
  const float* key   = (const float*)d_in[1];
  const float* value = (const float*)d_in[2];
  const float* Wq    = (const float*)d_in[3];
  const float* Wk    = (const float*)d_in[4];
  const float* Wv    = (const float*)d_in[5];
  const float* Wo    = (const float*)d_in[6];

  char* ws = (char*)d_ws;
  // bf16 intermediates: 16M elems (32MB) each
  unsigned short* qf  = (unsigned short*)(ws + 0);
  unsigned short* kf  = (unsigned short*)(ws + 33554432ull);
  unsigned short* vv  = (unsigned short*)(ws + 67108864ull);
  unsigned short* att = (unsigned short*)(ws + 100663296ull);
  float*          kvc = (float*)(ws + 134217728ull);  // [B*H, Nc, 64, 64] f32
  float*          zc  = (float*)(ws + 150994944ull);  // [B*H, Nc, 64] f32

  dim3 gg(DIM_ / 128, (B_ * L_) / 128), bb(256);
  proj_gemm<<<gg, bb, 0, stream>>>(query, Wq, qf, 1);
  proj_gemm<<<gg, bb, 0, stream>>>(key,   Wk, kf, 1);
  proj_gemm<<<gg, bb, 0, stream>>>(value, Wv, vv, 0);

  const int nchunks = B_ * H_ * NC_;                       // 1024
  size_t lds_kv   = (size_t)(CHUNK_*DH_ + DH_*CHUNK_) * 2; // 64KB
  chunk_kv<<<nchunks, 256, lds_kv, stream>>>(kf, vv, kvc, zc);

  prefix_scan<<<B_ * H_, 256, 0, stream>>>(kvc, zc);

  size_t lds_at = (size_t)(CHUNK_*DH_ + DH_*CHUNK_ + DH_*DH_) * 2
                + DH_ * 4 + 8 * 16 * 32 * 2;               // ~80.25KB (<320KB WGP)
  attn_chunk<<<nchunks, 256, lds_at, stream>>>(qf, kf, vv, kvc, zc, att);

  out_gemm<<<gg, bb, 0, stream>>>(att, Wo, (float*)d_out);
}
